// LinearAttention_46084999086236
// MI455X (gfx1250) — matrix operand order repaired and checked by reading, not compiled, not run on hardware
//
#include <hip/hip_runtime.h>
#include <hip/hip_bf16.h>

// ---------------------------------------------------------------------------
// Linear attention for MI455X (gfx1250, wave32, WMMA).
//
// Roofline: projections = 137 GFLOP vs ~300MB traffic (13us @ 23.3TB/s).
// bf16 WMMA 16x16x32 with hi/lo error compensation (3 wmma per K-step)
// gives near-fp32 accuracy at ~bf16 speed -> memory-bound.
// ---------------------------------------------------------------------------

typedef __attribute__((ext_vector_type(16))) __bf16 v16bf;
typedef __attribute__((ext_vector_type(8)))  __bf16 v8bf;
typedef __attribute__((ext_vector_type(4)))  __bf16 v4bf;
typedef __attribute__((ext_vector_type(8)))  float  v8f;

#define SEQ     4096
#define BATCH   4
#define DMODEL  1024
#define NHEADS  16
#define HDIM    64
#define MROWS   (SEQ * BATCH)          // 16384

// ---------------------------------------------------------------------------
// K0: split fp32 -> bf16 hi + bf16 lo  (a ~= hi + lo, error ~2^-16 rel)
// ---------------------------------------------------------------------------
__global__ __launch_bounds__(256) void split_bf16_k(const float* __restrict__ in,
                                                    __bf16* __restrict__ hi,
                                                    __bf16* __restrict__ lo,
                                                    int n4) {
    int i = blockIdx.x * 256 + threadIdx.x;
    if (i >= n4) return;
    float4 v = ((const float4*)in)[i];
    __bf16 h0 = (__bf16)v.x, h1 = (__bf16)v.y, h2 = (__bf16)v.z, h3 = (__bf16)v.w;
    v4bf H = {h0, h1, h2, h3};
    v4bf L = {(__bf16)(v.x - (float)h0), (__bf16)(v.y - (float)h1),
              (__bf16)(v.z - (float)h2), (__bf16)(v.w - (float)h3)};
    ((v4bf*)hi)[i] = H;
    ((v4bf*)lo)[i] = L;
}

// ---------------------------------------------------------------------------
// WMMA GEMM:  Out[m,n] = sum_k A[m,k] * W[n,k] + bias[n]   (A: MROWSxD, W: DxD)
// A,W given as split bf16 (hi/lo). Per wave: 16(M) x 64(N) tile, K-step 32.
// acc += Ah*Wh + Ah*Wl + Al*Wh   (fp32 accumulate via v_wmma_f32_16x16x32_bf16)
//
// act: 0 = none, 1 = elu(x)+1  (x>0 ? x+1 : exp(x))
// layout: 0 -> Out[m*1024+n] ([S,B,D]);  1 -> head-major Out[((b*16+h)*S+s)*64+d]
// ---------------------------------------------------------------------------
__global__ __launch_bounds__(128) void gemm_sb(const __bf16* __restrict__ Ah,
                                               const __bf16* __restrict__ Al,
                                               const __bf16* __restrict__ Wh,
                                               const __bf16* __restrict__ Wl,
                                               const float*  __restrict__ bias,
                                               float* __restrict__ out,
                                               int act, int layout) {
    const int lane = threadIdx.x & 31;
    const int wave = threadIdx.x >> 5;          // 4 waves, stacked along M
    const int lc   = lane & 15;
    const int kSel = lane >> 4;                  // 0 or 1 (K half owned by lane)
    const int mw    = blockIdx.x * 64 + wave * 16;   // wave's M base
    const int nBase = blockIdx.y * 64;               // WG's N base

    // A fragment rows (16-bit A, lanes hold M=lane%16; K = kSel*16 + e,
    // matching the B fragment's contiguous K order)
    const __bf16* arow_h = Ah + (size_t)(mw + lc) * DMODEL;
    const __bf16* arow_l = Al + (size_t)(mw + lc) * DMODEL;

    // B fragment rows: B(k,n)=W[n,k]; lane holds N=lane%16, K = kSel*16 + e
    const __bf16* brow_h[4];
    const __bf16* brow_l[4];
#pragma unroll
    for (int j = 0; j < 4; ++j) {
        size_t off = (size_t)(nBase + 16 * j + lc) * DMODEL + kSel * 16;
        brow_h[j] = Wh + off;
        brow_l[j] = Wl + off;
    }

    v8f acc[4] = {};

    for (int kc = 0; kc < DMODEL; kc += 32) {
        v16bf ah = *(const v16bf*)(arow_h + kc + kSel * 16);
        v16bf al = *(const v16bf*)(arow_l + kc + kSel * 16);

        v16bf bh[4], bl[4];
#pragma unroll
        for (int j = 0; j < 4; ++j) {
            bh[j] = *(const v16bf*)(brow_h[j] + kc);
            bl[j] = *(const v16bf*)(brow_l[j] + kc);
        }
        // three passes over 4 independent N-subtiles: hides wmma->wmma hazards
#pragma unroll
        for (int j = 0; j < 4; ++j)
            acc[j] = __builtin_amdgcn_wmma_f32_16x16x32_bf16(false, ah, false, bh[j],
                                                             (short)0, acc[j], false, false);
#pragma unroll
        for (int j = 0; j < 4; ++j)
            acc[j] = __builtin_amdgcn_wmma_f32_16x16x32_bf16(false, ah, false, bl[j],
                                                             (short)0, acc[j], false, false);
#pragma unroll
        for (int j = 0; j < 4; ++j)
            acc[j] = __builtin_amdgcn_wmma_f32_16x16x32_bf16(false, al, false, bh[j],
                                                             (short)0, acc[j], false, false);
    }

    // C/D layout: VGPR r -> M = r + 8*(lane>=16), N = lane%16
    const int rhi = kSel * 8;
#pragma unroll
    for (int j = 0; j < 4; ++j) {
        const int n = nBase + 16 * j + lc;
        const float bv = bias[n];
#pragma unroll
        for (int r = 0; r < 8; ++r) {
            const int m = mw + rhi + r;
            float v = acc[j][r] + bv;
            if (act) v = (v > 0.0f) ? (v + 1.0f) : __expf(v);
            if (layout == 0) {
                out[(size_t)m * DMODEL + n] = v;
            } else {
                const int s = m >> 2, bb = m & 3;        // m = s*BATCH + b
                const int hh = n >> 6, dd = n & 63;      // n = h*64 + d
                out[((((size_t)bb * NHEADS + hh) * SEQ + s) << 6) + dd] = v;
            }
        }
    }
}

// ---------------------------------------------------------------------------
// K2: per (b,h):  KV[d,v] = sum_s Kp[s,d]*V[s,v];  Ksum[d] = sum_s Kp[s,d]
// One WG per (b,h); LDS-staged 32-row chunks (coalesced b128 loads, broadcast
// LDS reads). Exact fp32 — this feeds the normalization denominator.
// ---------------------------------------------------------------------------
__global__ __launch_bounds__(256) void kv_ksum_k(const float* __restrict__ Kp,
                                                 const float* __restrict__ V,
                                                 float* __restrict__ KV,
                                                 float* __restrict__ Ksum) {
    __shared__ float sK[32 * HDIM];
    __shared__ float sV[32 * HDIM];
    const int bh = blockIdx.x;
    const int t  = threadIdx.x;
    const int d  = t & 63;
    const int vg = t >> 6;                  // 0..3 -> 16-wide v slab
    const float* Kb = Kp + (size_t)bh * SEQ * HDIM;
    const float* Vb = V  + (size_t)bh * SEQ * HDIM;

    float acc[16] = {};
    float ks = 0.0f;

    for (int s0 = 0; s0 < SEQ; s0 += 32) {
        __syncthreads();
        {   // 2048 floats each = 512 float4; 256 threads x 2
            const float4* srcK = (const float4*)(Kb + (size_t)s0 * HDIM);
            const float4* srcV = (const float4*)(Vb + (size_t)s0 * HDIM);
            ((float4*)sK)[t]       = srcK[t];
            ((float4*)sK)[t + 256] = srcK[t + 256];
            ((float4*)sV)[t]       = srcV[t];
            ((float4*)sV)[t + 256] = srcV[t + 256];
        }
        __syncthreads();
#pragma unroll 4
        for (int s = 0; s < 32; ++s) {
            const float kd = sK[s * HDIM + d];
            ks += kd;
            const float* vrow = sV + s * HDIM + vg * 16;
#pragma unroll
            for (int v = 0; v < 16; ++v) acc[v] += kd * vrow[v];
        }
    }

    float* KVb = KV + (size_t)bh * HDIM * HDIM;
#pragma unroll
    for (int v = 0; v < 16; ++v) KVb[d * HDIM + vg * 16 + v] = acc[v];
    if (vg == 0) Ksum[bh * HDIM + d] = ks;
}

// ---------------------------------------------------------------------------
// K3: per token:  z = Qp.Ksum + 1e-6;  y = (Qp @ KV)/z
// KV (16KB) cached in LDS, broadcast reads. Output written directly as
// split-bf16 in [S,B,D] layout -> ready for the final WMMA projection.
// ---------------------------------------------------------------------------
__global__ __launch_bounds__(256) void attn_apply_k(const float* __restrict__ Qp,
                                                    const float* __restrict__ KV,
                                                    const float* __restrict__ Ksum,
                                                    __bf16* __restrict__ Yh,
                                                    __bf16* __restrict__ Yl) {
    __shared__ float sKV[HDIM * HDIM];
    __shared__ float sKs[HDIM];
    const int bh = blockIdx.x;
    const int t  = threadIdx.x;
    {
        const float4* src = (const float4*)(KV + (size_t)bh * HDIM * HDIM);
#pragma unroll
        for (int i = 0; i < 4; ++i) ((float4*)sKV)[t + 256 * i] = src[t + 256 * i];
        if (t < HDIM) sKs[t] = Ksum[bh * HDIM + t];
    }
    __syncthreads();

    const int s = blockIdx.y * 256 + t;
    const float* qrow = Qp + ((size_t)bh * SEQ + s) * HDIM;
    float q[HDIM];
#pragma unroll
    for (int i = 0; i < 16; ++i) *(float4*)(q + 4 * i) = *(const float4*)(qrow + 4 * i);

    float z = 1e-6f;
#pragma unroll
    for (int d2 = 0; d2 < HDIM; ++d2) z += q[d2] * sKs[d2];
    const float invz = 1.0f / z;

    const int bb = bh >> 4, hh = bh & 15;
    const size_t mrow = ((size_t)s * BATCH + bb) * DMODEL + hh * HDIM;

    for (int vg = 0; vg < 4; ++vg) {
        float acc[16] = {};
        for (int d2 = 0; d2 < HDIM; ++d2) {
            const float qd = q[d2];
            const float* kv = sKV + d2 * HDIM + vg * 16;
#pragma unroll
            for (int v = 0; v < 16; ++v) acc[v] += qd * kv[v];
        }
#pragma unroll
        for (int v = 0; v < 16; ++v) {
            const float o = acc[v] * invz;
            const __bf16 oh = (__bf16)o;
            Yh[mrow + vg * 16 + v] = oh;
            Yl[mrow + vg * 16 + v] = (__bf16)(o - (float)oh);
        }
    }
}

// ---------------------------------------------------------------------------
extern "C" void kernel_launch(void* const* d_in, const int* in_sizes, int n_in,
                              void* d_out, int out_size, void* d_ws, size_t ws_size,
                              hipStream_t stream) {
    (void)in_sizes; (void)n_in; (void)out_size; (void)ws_size;
    const float* h  = (const float*)d_in[0];
    const float* x  = (const float*)d_in[1];
    const float* Wq = (const float*)d_in[2];
    const float* bq = (const float*)d_in[3];
    const float* Wk = (const float*)d_in[4];
    const float* bk = (const float*)d_in[5];
    const float* Wv = (const float*)d_in[6];
    const float* bv = (const float*)d_in[7];
    const float* Wo = (const float*)d_in[8];
    const float* bo = (const float*)d_in[9];
    float* out = (float*)d_out;
    char*  ws  = (char*)d_ws;

    const size_t MD2 = (size_t)MROWS * DMODEL * 2;   // bf16 activation plane
    const size_t DD2 = (size_t)DMODEL * DMODEL * 2;  // bf16 weight plane
    const size_t MD4 = (size_t)MROWS * DMODEL * 4;   // fp32 activation plane

    __bf16* h_hi = (__bf16*)(ws + 0 * MD2);
    __bf16* h_lo = (__bf16*)(ws + 1 * MD2);
    __bf16* x_hi = (__bf16*)(ws + 2 * MD2);
    __bf16* x_lo = (__bf16*)(ws + 3 * MD2);
    char* wp = ws + 4 * MD2;
    __bf16* Wq_hi = (__bf16*)(wp + 0 * DD2);
    __bf16* Wq_lo = (__bf16*)(wp + 1 * DD2);
    __bf16* Wk_hi = (__bf16*)(wp + 2 * DD2);
    __bf16* Wk_lo = (__bf16*)(wp + 3 * DD2);
    __bf16* Wv_hi = (__bf16*)(wp + 4 * DD2);
    __bf16* Wv_lo = (__bf16*)(wp + 5 * DD2);
    __bf16* Wo_hi = (__bf16*)(wp + 6 * DD2);
    __bf16* Wo_lo = (__bf16*)(wp + 7 * DD2);
    char* fp = wp + 8 * DD2;
    float* Qp  = (float*)(fp + 0 * MD4);             // head-major [B,H,S,64]
    float* Kp  = (float*)(fp + 1 * MD4);
    float* Vp  = (float*)(fp + 2 * MD4);
    float* KVb = (float*)(fp + 3 * MD4);             // [64, 64*64]
    float* Ks  = (float*)(fp + 3 * MD4 + (size_t)64 * HDIM * HDIM * 4);
    // Y overlays the h splits (dead after the Q projection)
    __bf16* Yh = (__bf16*)(ws + 0 * MD2);
    __bf16* Yl = (__bf16*)(ws + 1 * MD2);

    const int nMD4 = (MROWS * DMODEL) / 4;
    const int nDD4 = (DMODEL * DMODEL) / 4;
    split_bf16_k<<<(nMD4 + 255) / 256, 256, 0, stream>>>(h,  h_hi,  h_lo,  nMD4);
    split_bf16_k<<<(nMD4 + 255) / 256, 256, 0, stream>>>(x,  x_hi,  x_lo,  nMD4);
    split_bf16_k<<<(nDD4 + 255) / 256, 256, 0, stream>>>(Wq, Wq_hi, Wq_lo, nDD4);
    split_bf16_k<<<(nDD4 + 255) / 256, 256, 0, stream>>>(Wk, Wk_hi, Wk_lo, nDD4);
    split_bf16_k<<<(nDD4 + 255) / 256, 256, 0, stream>>>(Wv, Wv_hi, Wv_lo, nDD4);
    split_bf16_k<<<(nDD4 + 255) / 256, 256, 0, stream>>>(Wo, Wo_hi, Wo_lo, nDD4);

    dim3 gg(MROWS / 64, DMODEL / 64);                // 256 x 16 WGs, 128 thr
    gemm_sb<<<gg, 128, 0, stream>>>(h_hi, h_lo, Wq_hi, Wq_lo, bq, Qp, 1, 1);
    gemm_sb<<<gg, 128, 0, stream>>>(x_hi, x_lo, Wk_hi, Wk_lo, bk, Kp, 1, 1);
    gemm_sb<<<gg, 128, 0, stream>>>(x_hi, x_lo, Wv_hi, Wv_lo, bv, Vp, 0, 1);

    kv_ksum_k<<<BATCH * NHEADS, 256, 0, stream>>>(Kp, Vp, KVb, Ks);
    attn_apply_k<<<dim3(BATCH * NHEADS, SEQ / 256), 256, 0, stream>>>(Qp, KVb, Ks, Yh, Yl);

    gemm_sb<<<gg, 128, 0, stream>>>(Yh, Yl, Wo_hi, Wo_lo, bo, out, 0, 0);
}
